// DCNv4_pytorch_76338748719783
// MI455X (gfx1250) — matrix operand order repaired and checked by reading, compiled, not run on hardware
//
#include <hip/hip_runtime.h>
#include <hip/hip_bf16.h>
#include <cstdint>
#include <cstddef>

typedef __attribute__((ext_vector_type(16))) __bf16 v16bf;
typedef __attribute__((ext_vector_type(8)))  __bf16 v8bf;
typedef __attribute__((ext_vector_type(8)))  float  v8f;
typedef __attribute__((ext_vector_type(4)))  float  v4f;
typedef __attribute__((ext_vector_type(4)))  unsigned int u32x4;
typedef __attribute__((ext_vector_type(8)))  unsigned int u32x8;

namespace {
constexpr int kN  = 8;
constexpr int kC  = 256;
constexpr int kH  = 64;
constexpr int kW  = 64;
constexpr int kL  = kH * kW;        // 4096
constexpr int kM  = kN * kL;        // 32768 rows for all GEMMs
constexpr int kG  = 8;
constexpr int kP  = 9;
constexpr int kGC = kC / kG;        // 32 channels per group
constexpr int kOM = 216;            // G*P*3
}

// ---------------------------------------------------------------------------
// fp32 -> bf16 convert (weights)
// ---------------------------------------------------------------------------
__global__ void k_cvt_bf16(const float* __restrict__ in, __bf16* __restrict__ out, int n) {
    int i = blockIdx.x * blockDim.x + threadIdx.x;
    if (i < n) out[i] = (__bf16)in[i];
}

// ---------------------------------------------------------------------------
// x: NCHW fp32 -> NHWC bf16 (true transpose, feeds vp GEMM A)
// ---------------------------------------------------------------------------
__global__ void k_prep_x(const float* __restrict__ x, __bf16* __restrict__ xb) {
    int idx = blockIdx.x * blockDim.x + threadIdx.x;   // (n, l, c), c fastest
    int c = idx & (kC - 1);
    int l = (idx >> 8) & (kL - 1);
    int n = idx >> 20;
    xb[idx] = (__bf16)x[((size_t)n * kC + c) * kL + l];
}

// ---------------------------------------------------------------------------
// TDM: async-DMA one 64-row x 256-col bf16 B panel (row-major) into LDS.
// D# (CDNA5 ISA): group0 = {count, lds_addr, global_addr, type=2},
// group1 = 2-D descriptor: tensor_dim0=256 (K), tensor_dim0_stride=256,
// tensor_dim1=rowsAvail (rows past it read as zero -> handles Nout=216),
// tile = 256 x 64, data_size=2B. Groups 2/3 zero (tile_dim2=0 => unused).
// Issued by one wave; completion via s_wait_tensorcnt 0, then block barrier.
// ---------------------------------------------------------------------------
__device__ __forceinline__ void tdm_load_b_panel(const __bf16* gsrc,
                                                 unsigned rowsAvail,
                                                 unsigned ldsAddr) {
    unsigned long long ga = (unsigned long long)(uintptr_t)gsrc;
    u32x4 g0;
    g0[0] = 1u;                                        // count=1, no gather
    g0[1] = ldsAddr;                                   // LDS byte address
    g0[2] = (unsigned)(ga & 0xFFFFFFFFu);              // global_addr[31:0]
    g0[3] = ((unsigned)(ga >> 32) & 0x01FFFFFFu) | (2u << 30);  // addr[56:32] | type=2
    u32x8 g1;
    g1[0] = (1u << 16);          // workgroup_mask=0, data_size=1 (2 bytes)
    g1[1] = (256u << 16);        // tensor_dim0[15:0] = 256 @ bit48
    g1[2] = (rowsAvail << 16);   // tensor_dim1[15:0] @ bit80
    g1[3] = (256u << 16);        // tile_dim0 = 256 @ bit112
    g1[4] = 64u;                 // tile_dim1 = 64 rows
    g1[5] = 256u;                // tensor_dim0_stride = 256 elements
    g1[6] = 0u;
    g1[7] = 0u;
    u32x4 gz = {0u, 0u, 0u, 0u};
    asm volatile("tensor_load_to_lds %0, %1, %2, %3"
                 :: "s"(g0), "s"(g1), "s"(gz), "s"(gz)
                 : "memory");
    __builtin_amdgcn_s_wait_tensorcnt(0);
}

// ---------------------------------------------------------------------------
// Wave-level 16x64 WMMA strip (4 N-tiles share one A fragment per K-step).
// A: row-major M x K bf16 (global). sB: LDS B panel, 64 rows x 256 K.
// A-frag layout (ISA 7.12.2, 16-bit A 16x32): lane = m | (khalf<<4),
//   elems 0..7 = K kh..kh+7, elems 8..15 = K kh+16..kh+23, kh = (lane>>4)*8.
// B-frag: lane = n | (khalf<<4), same K order as A-frag (kh..kh+7, kh+16..kh+23).
// All 8 ds_load_b128 of a K-step are issued before the 4 WMMAs so they form
// one clause + a single dscnt wait (no per-WMMA LDS round-trip).
// ---------------------------------------------------------------------------
__device__ __forceinline__ void wmma_strip(const __bf16* __restrict__ A, int lda,
                                           const __bf16* sB, int m0, int n0,
                                           int nmax, int lane, v8f acc[4]) {
    const int ml  = m0 + (lane & 15);
    const int kh  = (lane >> 4) * 8;
    const int kbh = (lane >> 4) * 16;
    const __bf16* __restrict__ arow = A + (size_t)ml * lda;
    const __bf16* brow[4];
#pragma unroll
    for (int t = 0; t < 4; ++t) {
        int nl = n0 + t * 16 + (lane & 15);
        if (nl >= nmax) nl = nmax - 1;     // clamp; stores are guarded
        brow[t] = sB + (size_t)(nl - n0) * kC;
        v8f z = {};
        acc[t] = z;
    }
#pragma unroll
    for (int kb = 0; kb < kC; kb += 32) {
        union { v16bf v; v8bf h[2]; } ua;
        union { v16bf v; v8bf h[2]; } ub[4];
        ua.h[0] = *(const v8bf*)(arow + kb + kh);       // global_load_b128 x2
        ua.h[1] = *(const v8bf*)(arow + kb + 16 + kh);
#pragma unroll
        for (int t = 0; t < 4; ++t) {                   // ds_load_b128 x8 clause
            ub[t].h[0] = *(const v8bf*)(brow[t] + kb + kh);
            ub[t].h[1] = *(const v8bf*)(brow[t] + kb + 16 + kh);
        }
#pragma unroll
        for (int t = 0; t < 4; ++t) {                   // 4 back-to-back WMMAs
            acc[t] = __builtin_amdgcn_wmma_f32_16x16x32_bf16(
                         false, ua.v, false, ub[t].v, (short)0, acc[t], false, false);
        }
    }
}

// C/D layout: VGPR r -> row m0 + (lane>>4)*8 + r, col ntile + (lane&15).

// ---------------------------------------------------------------------------
// vp GEMM: v = x_nhwc @ vp_w^T + vp_b ; fp32 NHWC out (sampler needs fp32)
// ---------------------------------------------------------------------------
__global__ void k_gemm_vp(const __bf16* __restrict__ A, const __bf16* __restrict__ B,
                          const float* __restrict__ bias, float* __restrict__ C) {
    __shared__ __attribute__((aligned(16))) __bf16 sB[64 * kC];
    const int lane = threadIdx.x & 31;
    const int wave = threadIdx.x >> 5;
    const int n0 = (blockIdx.x & 3) * 64;
    const int m0 = ((blockIdx.x >> 2) * 8 + wave) * 16;
    if (wave == 0)
        tdm_load_b_panel(B + (size_t)n0 * kC, (unsigned)(kC - n0), (unsigned)(size_t)sB);
    __syncthreads();
    v8f acc[4];
    wmma_strip(A, kC, sB, m0, n0, kC, lane, acc);
    const int mBase = m0 + (lane >> 4) * 8;
#pragma unroll
    for (int t = 0; t < 4; ++t) {
        const int nOut = n0 + t * 16 + (lane & 15);
        const float b = bias[nOut];
#pragma unroll
        for (int r = 0; r < 8; ++r)
            C[(size_t)(mBase + r) * kC + nOut] = acc[t][r] + b;
    }
}

// ---------------------------------------------------------------------------
// depthwise 3x3 over raw-reshape NHWC view of x ; bf16 NHWC out (om GEMM A)
// ---------------------------------------------------------------------------
__global__ void k_dwconv(const float* __restrict__ x, const float* __restrict__ w,
                         const float* __restrict__ b, __bf16* __restrict__ out) {
    int idx = blockIdx.x * blockDim.x + threadIdx.x;     // (n,i,j,c), c fastest
    int c   = idx & (kC - 1);
    int pix = idx >> 8;
    int j   = pix & (kW - 1);
    int i   = (pix >> 6) & (kH - 1);
    int n   = pix >> 12;
    float acc = b[c];
#pragma unroll
    for (int ky = 0; ky < 3; ++ky) {
        int yy = i + ky - 1;
        if ((unsigned)yy >= (unsigned)kH) continue;
#pragma unroll
        for (int kx = 0; kx < 3; ++kx) {
            int xx = j + kx - 1;
            if ((unsigned)xx >= (unsigned)kW) continue;
            acc += x[(((size_t)n * kH + yy) * kW + xx) * kC + c] * w[c * 9 + ky * 3 + kx];
        }
    }
    out[idx] = (__bf16)acc;
}

// ---------------------------------------------------------------------------
// om GEMM: (32768 x 256) @ (256 x 216) + b ; fp32 out, compact stride 216.
// N covered by 4 groups of 64; TDM zero-fills rows >= 216, stores guarded.
// ---------------------------------------------------------------------------
__global__ void k_gemm_om(const __bf16* __restrict__ A, const __bf16* __restrict__ B,
                          const float* __restrict__ bias, float* __restrict__ C) {
    __shared__ __attribute__((aligned(16))) __bf16 sB[64 * kC];
    const int lane = threadIdx.x & 31;
    const int wave = threadIdx.x >> 5;
    const int n0 = (blockIdx.x & 3) * 64;
    const int m0 = ((blockIdx.x >> 2) * 8 + wave) * 16;
    if (wave == 0)
        tdm_load_b_panel(B + (size_t)n0 * kC, (unsigned)(kOM - n0), (unsigned)(size_t)sB);
    __syncthreads();
    v8f acc[4];
    wmma_strip(A, kC, sB, m0, n0, kOM, lane, acc);
    const int mBase = m0 + (lane >> 4) * 8;
#pragma unroll
    for (int t = 0; t < 4; ++t) {
        const int nOut = n0 + t * 16 + (lane & 15);
        if (nOut < kOM) {
            const float b = bias[nOut];
#pragma unroll
            for (int r = 0; r < 8; ++r)
                C[(size_t)(mBase + r) * kOM + nOut] = acc[t][r] + b;
        }
    }
}

// ---------------------------------------------------------------------------
// DCN bilinear sampler. One thread per (n, pixel, group); 32-ch accumulator.
// ix = w_o + dx + off_x, iy = h_o + dy + off_y over unpadded v, zero padding
// (equivalent to the reference's padded grid_sample path).
// ---------------------------------------------------------------------------
__global__ void k_dcn(const float* __restrict__ v, const float* __restrict__ om,
                      __bf16* __restrict__ out) {
    int t = blockIdx.x * blockDim.x + threadIdx.x;       // (n, l, g)
    int g = t & (kG - 1);
    int l = (t >> 3) & (kL - 1);
    int n = t >> 15;
    int wo = l & (kW - 1), ho = l >> 6;
    const float* __restrict__ omp = om + ((size_t)n * kL + l) * kOM + g * (3 * kP);
    const float* __restrict__ vb  = v + (size_t)n * kL * kC + g * kGC;

    float acc[kGC];
#pragma unroll
    for (int c = 0; c < kGC; ++c) acc[c] = 0.f;

    for (int p = 0; p < kP; ++p) {
        const int dx = p / 3 - 1;          // gx = repeat(wl,3)
        const int dy = p % 3 - 1;          // gy = tile(wl,3)
        const float offx = omp[2 * p];
        const float offy = omp[2 * p + 1];
        const float m    = omp[2 * kP + p];
        const float fx = (float)(wo + dx) + offx;
        const float fy = (float)(ho + dy) + offy;
        const float x0f = floorf(fx), y0f = floorf(fy);
        const float wx1 = fx - x0f, wx0 = 1.f - wx1;
        const float wy1 = fy - y0f, wy0 = 1.f - wy1;
        const int x0 = (int)x0f, y0 = (int)y0f;

        auto corner = [&](int yc, int xc, float wgt) {
            if ((unsigned)xc < (unsigned)kW && (unsigned)yc < (unsigned)kH) {
                const float* __restrict__ src = vb + ((size_t)(yc * kW + xc)) * kC;
                const float wm = wgt * m;
#pragma unroll
                for (int c = 0; c < kGC; c += 4) {
                    v4f val = *(const v4f*)(src + c);
                    acc[c + 0] += wm * val.x;
                    acc[c + 1] += wm * val.y;
                    acc[c + 2] += wm * val.z;
                    acc[c + 3] += wm * val.w;
                }
            }
        };
        corner(y0,     x0,     wy0 * wx0);
        corner(y0,     x0 + 1, wy0 * wx1);
        corner(y0 + 1, x0,     wy1 * wx0);
        corner(y0 + 1, x0 + 1, wy1 * wx1);
    }

    __bf16* __restrict__ dst = out + ((size_t)n * kL + l) * kC + g * kGC;
#pragma unroll
    for (int c = 0; c < kGC; ++c) dst[c] = (__bf16)acc[c];
}

// ---------------------------------------------------------------------------
// op GEMM: out = dcn @ op_w^T + op_b ; store transposed to NCHW fp32 d_out
// ---------------------------------------------------------------------------
__global__ void k_gemm_op(const __bf16* __restrict__ A, const __bf16* __restrict__ B,
                          const float* __restrict__ bias, float* __restrict__ C) {
    __shared__ __attribute__((aligned(16))) __bf16 sB[64 * kC];
    const int lane = threadIdx.x & 31;
    const int wave = threadIdx.x >> 5;
    const int n0 = (blockIdx.x & 3) * 64;
    const int m0 = ((blockIdx.x >> 2) * 8 + wave) * 16;
    if (wave == 0)
        tdm_load_b_panel(B + (size_t)n0 * kC, (unsigned)(kC - n0), (unsigned)(size_t)sB);
    __syncthreads();
    v8f acc[4];
    wmma_strip(A, kC, sB, m0, n0, kC, lane, acc);
    const int mBase = m0 + (lane >> 4) * 8;
#pragma unroll
    for (int t = 0; t < 4; ++t) {
        const int nOut = n0 + t * 16 + (lane & 15);
        const float b = bias[nOut];
#pragma unroll
        for (int r = 0; r < 8; ++r) {
            const int mIdx = mBase + r;
            const int n = mIdx >> 12;           // / kL
            const int l = mIdx & (kL - 1);
            C[((size_t)n * kC + nOut) * kL + l] = acc[t][r] + b;
        }
    }
}

// ---------------------------------------------------------------------------
extern "C" void kernel_launch(void* const* d_in, const int* in_sizes, int n_in,
                              void* d_out, int out_size, void* d_ws, size_t ws_size,
                              hipStream_t stream) {
    const float* x    = (const float*)d_in[0];
    const float* dw_w = (const float*)d_in[1];
    const float* dw_b = (const float*)d_in[2];
    const float* om_w = (const float*)d_in[3];
    const float* om_b = (const float*)d_in[4];
    const float* vp_w = (const float*)d_in[5];
    const float* vp_b = (const float*)d_in[6];
    const float* op_w = (const float*)d_in[7];
    const float* op_b = (const float*)d_in[8];
    float* out = (float*)d_out;

    // workspace carve-out (all 256B aligned)
    char* ws = (char*)d_ws;
    size_t off = 0;
    auto carve = [&](size_t bytes) {
        void* p = ws + off;
        off = (off + bytes + 255) & ~(size_t)255;
        return p;
    };
    __bf16* xb    = (__bf16*)carve((size_t)kM * kC * 2);     // x NHWC bf16
    __bf16* vpwb  = (__bf16*)carve((size_t)kC * kC * 2);
    __bf16* omwb  = (__bf16*)carve((size_t)kOM * kC * 2);
    __bf16* opwb  = (__bf16*)carve((size_t)kC * kC * 2);
    float*  vf    = (float*)carve((size_t)kM * kC * 4);      // v NHWC fp32
    __bf16* dwb   = (__bf16*)carve((size_t)kM * kC * 2);     // dw NHWC bf16
    float*  omf   = (float*)carve((size_t)kM * kOM * 4);     // offsets+mask
    __bf16* dcnb  = (__bf16*)carve((size_t)kM * kC * 2);     // dcn NHWC bf16

    // weights -> bf16
    k_cvt_bf16<<<(kC * kC + 255) / 256, 256, 0, stream>>>(vp_w, vpwb, kC * kC);
    k_cvt_bf16<<<(kOM * kC + 255) / 256, 256, 0, stream>>>(om_w, omwb, kOM * kC);
    k_cvt_bf16<<<(kC * kC + 255) / 256, 256, 0, stream>>>(op_w, opwb, kC * kC);

    // x transpose to NHWC bf16
    k_prep_x<<<(kM * kC) / 256, 256, 0, stream>>>(x, xb);

    // vp GEMM: 256 m-groups x 4 n-groups
    k_gemm_vp<<<(kM / 16 / 8) * 4, 256, 0, stream>>>(xb, vpwb, vp_b, vf);

    // depthwise conv
    k_dwconv<<<(kM * kC) / 256, 256, 0, stream>>>(x, dw_w, dw_b, dwb);

    // om GEMM
    k_gemm_om<<<(kM / 16 / 8) * 4, 256, 0, stream>>>(dwb, omwb, om_b, omf);

    // deformable sampling
    k_dcn<<<(kM * kG) / 256, 256, 0, stream>>>(vf, omf, dcnb);

    // op GEMM -> NCHW fp32 output
    k_gemm_op<<<(kM / 16 / 8) * 4, 256, 0, stream>>>(dcnb, opwb, op_b, out);
}
